// linear_sub_28776280883572
// MI455X (gfx1250) — matrix operand order repaired and checked by reading, not compiled, not run on hardware
//
#include <hip/hip_runtime.h>
#include <hip/hip_bf16.h>
#include <stdint.h>

#define DIM 512

typedef __attribute__((ext_vector_type(16))) __bf16  v16bf;
typedef __attribute__((ext_vector_type(8)))  float   v8f;
typedef __attribute__((ext_vector_type(4)))  float   f32x4;
typedef __attribute__((ext_vector_type(4)))  unsigned int u32x4;
typedef __attribute__((ext_vector_type(2)))  unsigned int u32x2;

// Padded LDS row stride (ushorts): 512 + 8 spreads lanes across banks.
#define LROW 520

__device__ __forceinline__ unsigned short f32_to_bf16_rne(float f) {
  uint32_t u = __builtin_bit_cast(uint32_t, f);
  uint32_t r = u + 0x7FFFu + ((u >> 16) & 1u);
  return (unsigned short)(r >> 16);
}
__device__ __forceinline__ float bf16_bits_to_f32(unsigned short h) {
  uint32_t u = ((uint32_t)h) << 16;
  return __builtin_bit_cast(float, u);
}

// Load 16 bf16 (two 16-byte chunks at p+off and p+off+step ushorts) into a v16bf.
__device__ __forceinline__ v16bf ld_pair(const unsigned short* p, int off, int step) {
  union { u32x4 u[2]; v16bf v; } t;
  t.u[0] = *(const u32x4*)(p + off);
  t.u[1] = *(const u32x4*)(p + off + step);
  return t.v;
}

// ---------------------------------------------------------------------------
// Prep: S = A + A^T, split into bf16 hi/lo planes in workspace.
// ---------------------------------------------------------------------------
__global__ __launch_bounds__(256) void prep_S(const float* __restrict__ A,
                                              unsigned short* __restrict__ Shi,
                                              unsigned short* __restrict__ Slo) {
  int idx = blockIdx.x * 256 + threadIdx.x;          // 0 .. 512*512-1
  int r = idx >> 9, c = idx & (DIM - 1);
  float s = A[r * DIM + c] + A[c * DIM + r];
  unsigned short hb = f32_to_bf16_rne(s);
  float rem = s - bf16_bits_to_f32(hb);
  Shi[idx] = hb;
  Slo[idx] = f32_to_bf16_rne(rem);
}

// ---------------------------------------------------------------------------
// Main: out[:, :512] = p + q*S (bf16x3 WMMA), out[:, 512:] = q
// Workgroup: 256 threads (8 waves), 128 rows of q. Grid: 512 blocks.
// ---------------------------------------------------------------------------
__global__ __launch_bounds__(256) void fused_qS(const float* __restrict__ pq,
                                                const unsigned short* __restrict__ Shi,
                                                const unsigned short* __restrict__ Slo,
                                                float* __restrict__ out) {
  extern __shared__ unsigned short smem[];
  unsigned short* qhi = smem;                       // 128 x LROW
  unsigned short* qlo = qhi + 128 * LROW;           // 128 x LROW
  unsigned short* shl = qlo + 128 * LROW;           // 16 x LROW (S_hi stripe)
  unsigned short* sll = shl + 16 * LROW;            // 16 x LROW (S_lo stripe)

  const int tid  = threadIdx.x;
  const int wave = tid >> 5;
  const int lane = tid & 31;
  const int row0 = blockIdx.x * 128;

  // ---- Phase 0: load q block, emit q pass-through, stage bf16 hi/lo in LDS.
  const f32x4* pq4  = (const f32x4*)pq;
  f32x4*       out4 = (f32x4*)out;
  for (int i4 = tid; i4 < 128 * 128; i4 += 256) {   // 128 rows x 128 float4/row
    int row = i4 >> 7;
    int c4  = i4 & 127;
    long gi = (long)(row0 + row) * 256 + 128 + c4;  // q half, float4 units
    f32x4 v = pq4[gi];
    out4[gi] = v;                                   // out[:,512:] = q
    union { unsigned short s[4]; u32x2 v; } ph, pl;
    #pragma unroll
    for (int j = 0; j < 4; ++j) {
      float f = v[j];
      unsigned short hb = f32_to_bf16_rne(f);
      ph.s[j] = hb;
      pl.s[j] = f32_to_bf16_rne(f - bf16_bits_to_f32(hb));
    }
    *(u32x2*)&qhi[row * LROW + c4 * 4] = ph.v;
    *(u32x2*)&qlo[row * LROW + c4 * 4] = pl.v;
  }

  // Per-wave constant addressing for WMMA operands.
  const int mrow  = (wave << 4) + (lane & 15);
  const int abase = mrow * LROW;
  const int ahalf = (lane >> 4) << 3;               // 0 or 8 (A chunk select)
  const int bbase = (lane & 15) * LROW + ((lane >> 4) << 3); // 0 or 8 (B chunk select)

  for (int nt = 0; nt < 32; ++nt) {
    __syncthreads();                                // prev compute done
    // ---- Stage 16x512 stripe of S_hi/S_lo (rows e0..e0+15; symmetric: row==col)
    const int e0 = nt << 4;
    for (int j = tid; j < 1024; j += 256) {         // 16 rows x 64 u32x4
      int r = j >> 6, c = j & 63;
      *(u32x4*)&shl[r * LROW + c * 8] = ((const u32x4*)(Shi + (e0 + r) * DIM))[c];
      *(u32x4*)&sll[r * LROW + c * 8] = ((const u32x4*)(Slo + (e0 + r) * DIM))[c];
    }
    __syncthreads();

    // ---- K loop: 16 slabs of 32, bf16x3 per slab.
    v8f acc = {};
    #pragma unroll 4
    for (int sl = 0; sl < 16; ++sl) {
      const int k0 = sl << 5;
      v16bf ahi = ld_pair(qhi, abase + k0 + ahalf, 16);
      v16bf alo = ld_pair(qlo, abase + k0 + ahalf, 16);
      v16bf bhi = ld_pair(shl, bbase + k0, 16);
      v16bf blo = ld_pair(sll, bbase + k0, 16);
      acc = __builtin_amdgcn_wmma_f32_16x16x32_bf16(false, ahi, false, bhi,
                                                    (short)0, acc, false, false);
      acc = __builtin_amdgcn_wmma_f32_16x16x32_bf16(false, ahi, false, blo,
                                                    (short)0, acc, false, false);
      acc = __builtin_amdgcn_wmma_f32_16x16x32_bf16(false, alo, false, bhi,
                                                    (short)0, acc, false, false);
    }

    // ---- Store: out[:, :512] = p + acc. D layout: VGPR r -> M=r(+8 for hi lanes).
    const int col   = e0 + (lane & 15);
    const int rbase = row0 + (wave << 4) + ((lane >> 4) << 3);
    #pragma unroll
    for (int r = 0; r < 8; ++r) {
      long gi = (long)(rbase + r) * 1024 + col;
      out[gi] = pq[gi] + acc[r];
    }
  }
}

// ---------------------------------------------------------------------------
extern "C" void kernel_launch(void* const* d_in, const int* in_sizes, int n_in,
                              void* d_out, int out_size, void* d_ws, size_t ws_size,
                              hipStream_t stream) {
  const float* pq = (const float*)d_in[0];          // 65536 x 1024 fp32
  const float* A  = (const float*)d_in[1];          // 512 x 512 fp32
  float* out      = (float*)d_out;                  // 65536 x 1024 fp32

  unsigned short* Shi = (unsigned short*)d_ws;      // 512 KB
  unsigned short* Slo = Shi + DIM * DIM;            // 512 KB

  prep_S<<<(DIM * DIM) / 256, 256, 0, stream>>>(A, Shi, Slo);

  const size_t lds_bytes = (size_t)(2 * 128 * LROW + 2 * 16 * LROW) * sizeof(unsigned short);
  fused_qS<<<512, 256, lds_bytes, stream>>>(pq, Shi, Slo, out);
}
